// ITDA_37821482009380
// MI455X (gfx1250) — matrix operand order repaired and checked by reading, not compiled, not run on hardware
//
#include <hip/hip_runtime.h>
#include <stdint.h>

#define BATCH   2048
#define DICT    4096
#define DIM     1024
#define KSTEPS  8
#define KCHUNKS (DIM / 32)   // 32 k-chunks of 32 bf16

// Block-cooperative GEMM tiling
#define BM   128             // A rows per workgroup tile
#define BN   256             // B rows per workgroup tile
#define AST  40              // padded LDS row stride (bf16 elems): 20 dwords -> conflict-free b128 reads

typedef __attribute__((ext_vector_type(16))) __bf16 v16bf;
typedef __attribute__((ext_vector_type(8)))  float  v8f;

union Frag32B {
    uint4 u[2];
    v16bf v;
};

static __device__ __forceinline__ unsigned short f32_to_bf16_rne(float f) {
    unsigned u = __float_as_uint(f);
    unsigned r = u + 0x7FFFu + ((u >> 16) & 1u);
    return (unsigned short)(r >> 16);
}
static __device__ __forceinline__ float bf16_to_f32(unsigned short h) {
    return __uint_as_float(((unsigned)h) << 16);
}

// ---- CDNA5 async global->LDS copy (ASYNCcnt-tracked, no VGPR round-trip) ----
static __device__ __forceinline__ void async_load16(unsigned ldsOff, const void* gptr) {
    asm volatile("global_load_async_to_lds_b128 %0, %1, off"
                 :: "v"(ldsOff), "v"(gptr) : "memory");
}
static __device__ __forceinline__ void wait_async0() {
#if __has_builtin(__builtin_amdgcn_s_wait_asynccnt)
    __builtin_amdgcn_s_wait_asynccnt(0);
#else
    asm volatile("s_wait_asynccnt 0" ::: "memory");
#endif
}

// ---------------------------------------------------------------------------
// f32 -> bf16 conversion (4 elements / thread)
// ---------------------------------------------------------------------------
__global__ __launch_bounds__(256) void cvt_bf16(const float* __restrict__ src,
                                                unsigned short* __restrict__ dst) {
    int i = (blockIdx.x * 256 + threadIdx.x) * 4;
    float4 f = *(const float4*)(src + i);
    ushort4 o;
    o.x = f32_to_bf16_rne(f.x);
    o.y = f32_to_bf16_rne(f.y);
    o.z = f32_to_bf16_rne(f.z);
    o.w = f32_to_bf16_rne(f.w);
    *(ushort4*)(dst + i) = o;
}

// ---------------------------------------------------------------------------
// Stage one k-chunk: A tile 128x32 (8 KB) + B tile 256x32 (16 KB) via async
// b128 copies; 256 threads issue 2 + 4 ops each.
// ---------------------------------------------------------------------------
static __device__ __forceinline__ void stage_tile(const unsigned short* __restrict__ Abase,
                                                  const unsigned short* __restrict__ Bbase,
                                                  unsigned short* As, unsigned short* Bs,
                                                  int kofs, int tid) {
#pragma unroll
    for (int p = 0; p < 2; ++p) {           // 512 x 16B chunks of A
        int c = tid + 256 * p;
        int row = c >> 2, q = c & 3;
        async_load16((unsigned)(uintptr_t)(As + row * AST + q * 8),
                     Abase + (size_t)row * DIM + kofs + q * 8);
    }
#pragma unroll
    for (int p = 0; p < 4; ++p) {           // 1024 x 16B chunks of B
        int c = tid + 256 * p;
        int row = c >> 2, q = c & 3;
        async_load16((unsigned)(uintptr_t)(Bs + row * AST + q * 8),
                     Bbase + (size_t)row * DIM + kofs + q * 8);
    }
}

// ---------------------------------------------------------------------------
// Per-wave 64x64 sub-tile compute for one staged 32-k chunk (16 WMMAs)
// ---------------------------------------------------------------------------
static __device__ __forceinline__ void compute_chunk(const unsigned short* __restrict__ As,
                                                     const unsigned short* __restrict__ Bs,
                                                     int waveM, int waveN, int nlo, int hi,
                                                     v8f acc[4][4]) {
    Frag32B a[4], b[4];
#pragma unroll
    for (int t = 0; t < 4; ++t) {
        // A frag (16x32 bf16): half 0 -> K 0-7 & 16-23, half 1 -> K 8-15 & 24-31
        const uint4* pa = (const uint4*)(As + (waveM * 64 + t * 16 + nlo) * AST + hi * 8);
        a[t].u[0] = pa[0];
        a[t].u[1] = pa[2];
        // B frag (32x16 bf16): half 0 -> K 0-7 & 16-23, half 1 -> K 8-15 & 24-31
        const uint4* pb = (const uint4*)(Bs + (waveN * 64 + t * 16 + nlo) * AST + hi * 8);
        b[t].u[0] = pb[0];
        b[t].u[1] = pb[2];
    }
#pragma unroll
    for (int ti = 0; ti < 4; ++ti)
#pragma unroll
        for (int tj = 0; tj < 4; ++tj)
            acc[ti][tj] = __builtin_amdgcn_wmma_f32_16x16x32_bf16(
                false, a[ti].v, false, b[tj].v, (short)0, acc[ti][tj], false, false);
}

// ---------------------------------------------------------------------------
// C[i,j] = Arows[i] . Brows[j]   (row-major bf16 inputs, C is [M x DICT])
// WG tile 128x256, waves 2x4, double-buffered async LDS staging.
// ---------------------------------------------------------------------------
template <bool OUT_BF16>
__global__ __launch_bounds__(256) void gemm_nt_kernel(const unsigned short* __restrict__ Arows,
                                                      const unsigned short* __restrict__ Brows,
                                                      void* __restrict__ Cout) {
    __shared__ unsigned short As[2][BM * AST];   // 2 x 10 KB
    __shared__ unsigned short Bs[2][BN * AST];   // 2 x 20 KB

    const int tid = threadIdx.x;
    const int lane = tid & 31, wave = tid >> 5;
    const int waveM = wave >> 2, waveN = wave & 3;
    const int nlo = lane & 15, hi = (lane >> 4) & 1;

    const int iWG = (blockIdx.x >> 4) * BM;      // DICT/BN == 16 column blocks
    const int jWG = (blockIdx.x & 15) * BN;
    const unsigned short* Abase = Arows + (size_t)iWG * DIM;
    const unsigned short* Bbase = Brows + (size_t)jWG * DIM;

    v8f acc[4][4];
#pragma unroll
    for (int ti = 0; ti < 4; ++ti)
#pragma unroll
        for (int tj = 0; tj < 4; ++tj)
            acc[ti][tj] = v8f{0.f, 0.f, 0.f, 0.f, 0.f, 0.f, 0.f, 0.f};

    stage_tile(Abase, Bbase, As[0], Bs[0], 0, tid);
    wait_async0();
    __syncthreads();

    for (int kc = 0; kc < KCHUNKS; ++kc) {
        const int cur = kc & 1;
        if (kc + 1 < KCHUNKS)
            stage_tile(Abase, Bbase, As[cur ^ 1], Bs[cur ^ 1], (kc + 1) * 32, tid);
        compute_chunk(As[cur], Bs[cur], waveM, waveN, nlo, hi, acc);
        if (kc + 1 < KCHUNKS) wait_async0();
        __syncthreads();   // stage(k+1) done everywhere AND buffer cur fully consumed
    }

    const int lane2 = tid & 31;
    const int nl = lane2 & 15, h2 = (lane2 >> 4) & 1;
#pragma unroll
    for (int ti = 0; ti < 4; ++ti)
#pragma unroll
        for (int tj = 0; tj < 4; ++tj)
#pragma unroll
            for (int j = 0; j < 8; ++j) {
                // C/D layout: VGPR j -> row j (lanes 0-15) / row j+8 (lanes 16-31)
                int row = iWG + waveM * 64 + ti * 16 + j + h2 * 8;
                int col = jWG + waveN * 64 + tj * 16 + nl;
                if constexpr (OUT_BF16)
                    ((unsigned short*)Cout)[(size_t)row * DICT + col] = f32_to_bf16_rne(acc[ti][tj][j]);
                else
                    ((float*)Cout)[(size_t)row * DICT + col] = acc[ti][tj][j];
            }
}

// ---------------------------------------------------------------------------
// Fused 8-step matching pursuit, fully row-local (corr rows live in LDS):
//   per step/row: argmax |corr| -> exact f32 val = R.atoms[best]
//                 -> R -= val*atoms[best] ; corr -= val*G[best,:]
// Finally out = x - R  (== acts @ atoms).
// ---------------------------------------------------------------------------
#define MPROWS 2
__global__ __launch_bounds__(256) void mp_steps_kernel(const float* __restrict__ x,
                                                       const float* __restrict__ atomsF,
                                                       const unsigned short* __restrict__ G,
                                                       const float* __restrict__ corrG,
                                                       float* __restrict__ out) {
    __shared__ float corrL[MPROWS * DICT];            // 32 KB
    __shared__ float Rl[MPROWS * DIM];                // 8 KB
    __shared__ unsigned long long keys[MPROWS];       // (absbits<<32) | ~col
    __shared__ float vals[MPROWS];
    __shared__ int   bidx[MPROWS];

    const int tid = threadIdx.x, lane = tid & 31, wave = tid >> 5;
    const int rbase = blockIdx.x * MPROWS;

    for (int i = tid; i < MPROWS * DICT; i += 256) corrL[i] = corrG[(size_t)rbase * DICT + i];
    for (int i = tid; i < MPROWS * DIM; i += 256)  Rl[i]    = x[(size_t)rbase * DIM + i];
    __syncthreads();

    for (int s = 0; s < KSTEPS; ++s) {
        if (tid < MPROWS) keys[tid] = 0x00000000FFFFFFFFull;   // abs=0, col=0
        __syncthreads();

        {   // argmax: 128 threads per row, then ds_max_u64 combine (~col => first-index ties)
            const int row = tid >> 7;
            float bestA = 0.0f; int bestC = 0;
            for (int c = (tid & 127); c < DICT; c += 128) {
                float v = fabsf(corrL[row * DICT + c]);
                if (v > bestA) { bestA = v; bestC = c; }
            }
            unsigned long long key = ((unsigned long long)__float_as_uint(bestA) << 32) |
                                     (unsigned long long)(0xFFFFFFFFu - (unsigned)bestC);
            atomicMax(&keys[row], key);
        }
        __syncthreads();

        if (wave < MPROWS) {   // exact f32 coefficient for the selected atom
            const int row = wave;
            unsigned idx = 0xFFFFFFFFu - (unsigned)(keys[row] & 0xFFFFFFFFull);
            const float* ap = atomsF + (size_t)idx * DIM;
            float sum = 0.0f;
            for (int d = lane; d < DIM; d += 32) sum += Rl[row * DIM + d] * ap[d];
#pragma unroll
            for (int off = 16; off > 0; off >>= 1) sum += __shfl_xor(sum, off, 32);
            if (lane == 0) { vals[row] = sum; bidx[row] = (int)idx; }
        }
        __syncthreads();

        for (int i = tid; i < MPROWS * DIM; i += 256) {        // residual update (f32)
            int row = i >> 10, col = i & (DIM - 1);
            Rl[i] -= vals[row] * atomsF[(size_t)bidx[row] * DIM + col];
        }
        for (int i = tid; i < MPROWS * DICT; i += 256) {       // corr update via Gram row
            int row = i >> 12, col = i & (DICT - 1);
            corrL[i] -= vals[row] * bf16_to_f32(G[(size_t)bidx[row] * DICT + col]);
        }
        __syncthreads();
    }

    for (int i = tid; i < MPROWS * DIM; i += 256) {            // decode: out = x - R
        size_t g = (size_t)rbase * DIM + i;
        out[g] = x[g] - Rl[i];
    }
}

extern "C" void kernel_launch(void* const* d_in, const int* in_sizes, int n_in,
                              void* d_out, int out_size, void* d_ws, size_t ws_size,
                              hipStream_t stream) {
    (void)in_sizes; (void)n_in; (void)out_size; (void)ws_size;
    const float* x     = (const float*)d_in[0];
    const float* atoms = (const float*)d_in[1];
    // d_in[2] is k (== 8), baked in as KSTEPS.

    char* ws = (char*)d_ws;
    unsigned short* atomsB = (unsigned short*)ws;                               // 8 MB
    unsigned short* xB     = (unsigned short*)(ws + (size_t)DICT * DIM * 2);    // 4 MB
    unsigned short* G      = (unsigned short*)(ws + (size_t)DICT * DIM * 2
                                                  + (size_t)BATCH * DIM * 2);   // 32 MB
    float* corr            = (float*)(ws + (size_t)DICT * DIM * 2
                                         + (size_t)BATCH * DIM * 2
                                         + (size_t)DICT * DICT * 2);            // 32 MB

    cvt_bf16<<<(DICT * DIM) / 1024, 256, 0, stream>>>(atoms, atomsB);
    cvt_bf16<<<(BATCH * DIM) / 1024, 256, 0, stream>>>(x, xB);
    // Gram: 4096x4096 from atoms (bf16 out) — (4096/128)*(4096/256) = 512 blocks
    gemm_nt_kernel<true><<<(DICT / BM) * (DICT / BN), 256, 0, stream>>>(atomsB, atomsB, G);
    // corr0: 2048x4096 from x (f32 out) — (2048/128)*(4096/256) = 256 blocks
    gemm_nt_kernel<false><<<(BATCH / BM) * (DICT / BN), 256, 0, stream>>>(xB, atomsB, corr);
    mp_steps_kernel<<<BATCH / MPROWS, 256, 0, stream>>>(x, atoms, G, corr, (float*)d_out);
}
